// TVAEDecoder_46815143526984
// MI455X (gfx1250) — matrix operand order repaired and checked by reading, not compiled, not run on hardware
//
#include <hip/hip_runtime.h>
#include <math.h>

typedef __bf16 bf16;
typedef __attribute__((ext_vector_type(8)))  __bf16 v8bf;
typedef __attribute__((ext_vector_type(16))) __bf16 v16bf;
typedef __attribute__((ext_vector_type(8)))  float  v8f;

#define AS1 __attribute__((address_space(1)))
typedef const AS1 bf16* gwptr;

// ---------------- problem constants ----------------
constexpr int T_STEPS = 254;   // len(states[:-2])
constexpr int BATCH   = 512;
constexpr int STATE   = 32;
constexpr int ZDIM    = 64;
constexpr int RNN     = 512;
constexpr int FCIN    = 608;   // 32 + 64 + 512
constexpr float LOG2PI_F = 1.8378770664093453f;

// ---------------- bf16 weight layout in d_ws (element offsets) ----------------
constexpr size_t O_WIH0  = 0;
constexpr size_t O_WHH0  = O_WIH0  + (size_t)1536 * 64;
constexpr size_t O_WIH1  = O_WHH0  + (size_t)1536 * 512;
constexpr size_t O_WHH1  = O_WIH1  + (size_t)1536 * 512;
constexpr size_t O_WFC1  = O_WHH1  + (size_t)1536 * 512;
constexpr size_t O_WFC2  = O_WFC1  + (size_t)512 * 608;
constexpr size_t O_WMEAN = O_WFC2  + (size_t)512 * 512;
constexpr size_t O_WLV   = O_WMEAN + (size_t)32 * 512;

// ---------------- helpers ----------------
__device__ __forceinline__ v16bf cat8(v8bf lo, v8bf hi) {
  return __builtin_shufflevector(lo, hi, 0,1,2,3,4,5,6,7,8,9,10,11,12,13,14,15);
}

// Opaque pass-through (SGPR pair) to defeat LICM on weight loads, with an
// explicit global-addrspace result so the loads lower to global_load_b128
// (LOADcnt only) instead of flat_load_b128 (LOADcnt + DScnt coupling).
__device__ __forceinline__ gwptr opq(const bf16* p) {
  unsigned long long u = (unsigned long long)p;
  asm volatile("" : "+s"(u));
  return (gwptr)u;
}

// A-fragment (16x32, bf16) from a row-major LDS tile [16][rowStride].
// Lane L: row = L%16; halves K = {k0..k0+7, k0+16..k0+23}, +8 for lanes>=16.
__device__ __forceinline__ v16bf load_a_lds(const bf16* base, int rowStride, int k0, int lane) {
  const bf16* p = base + (lane & 15) * rowStride + k0 + ((lane >> 4) << 3);
  v8bf lo = *reinterpret_cast<const v8bf*>(p);
  v8bf hi = *reinterpret_cast<const v8bf*>(p + 16);
  return cat8(lo, hi);
}

// B-fragment (32x16) for D = A x W^T: B[k][n] = W[n][k] (W row-major (N,K) bf16).
// Lane L: n = n0 + L%16; K = {k0..k0+7, k0+16..k0+23}, +8 for lanes>=16.
// 32-bit element offset (max ~786K) so ISel can use the GVS addressing mode:
// SGPR64 base + one VGPR 32-bit offset, instead of a per-lane 64-bit vaddr pair.
__device__ __forceinline__ v16bf load_b_w(gwptr W, int K, int n0, int k0, int lane) {
  const int off = (n0 + (lane & 15)) * K + k0 + ((lane >> 4) << 3);
  gwptr p = W + off;
  v8bf lo = *reinterpret_cast<const AS1 v8bf*>(p);
  v8bf hi = *reinterpret_cast<const AS1 v8bf*>(p + 16);
  return cat8(lo, hi);
}

__device__ __forceinline__ v8f wmma_bf16(v16bf a, v16bf b, v8f c) {
  return __builtin_amdgcn_wmma_f32_16x16x32_bf16(false, a, false, b, (short)0, c, false, false);
}

__device__ __forceinline__ v8f z8() {
  v8f v = {0.f, 0.f, 0.f, 0.f, 0.f, 0.f, 0.f, 0.f};
  return v;
}

__device__ __forceinline__ float sigf(float x) { return 1.0f / (1.0f + __expf(-x)); }

// ---------------- weight f32 -> bf16 conversion ----------------
__global__ void cvt_f32_to_bf16(const float* __restrict__ src, bf16* __restrict__ dst, int n) {
  int i = blockIdx.x * blockDim.x + threadIdx.x;
  if (i < n) dst[i] = (bf16)src[i];
}

// ---------------- main persistent kernel ----------------
// 32 workgroups x 512 threads (16 wave32). Workgroup g owns batch rows [16g, 16g+16).
__global__ __launch_bounds__(512, 1)
void tvae_decoder_kernel(
    const float* __restrict__ states, const float* __restrict__ z,
    const bf16* __restrict__ Wih0p, const bf16* __restrict__ Whh0p,
    const float* __restrict__ bih0, const float* __restrict__ bhh0,
    const bf16* __restrict__ Wih1p, const bf16* __restrict__ Whh1p,
    const float* __restrict__ bih1, const float* __restrict__ bhh1,
    const bf16* __restrict__ Wfc1p, const float* __restrict__ bfc1,
    const bf16* __restrict__ Wfc2p, const float* __restrict__ bfc2,
    const bf16* __restrict__ Wmeanp, const float* __restrict__ bmean,
    const bf16* __restrict__ Wlvp,  const float* __restrict__ blv,
    float* __restrict__ out)
{
  __shared__ bf16  s_bf [16 * STATE];   // 1 KB  current state tile
  __shared__ bf16  a_bf [16 * STATE];   // 1 KB  action tile (bf16 for WMMA)
  __shared__ float a_f32[16 * STATE];   // 2 KB  action tile (f32 for NLL)
  __shared__ bf16  z_bf [16 * ZDIM];    // 2 KB  latent tile (invariant)
  __shared__ bf16  h0_bf[16 * RNN];     // 16 KB GRU layer-0 hidden
  __shared__ bf16  h1_bf[16 * RNN];     // 16 KB GRU layer-1 hidden
  __shared__ bf16  dh_bf[16 * RNN];     // 16 KB MLP activation (reused fc1->fc2)
  __shared__ float mean_ls[16 * STATE]; // 2 KB
  __shared__ float lv_ls  [16 * STATE]; // 2 KB
  __shared__ float nll_ls [16 * STATE]; // 2 KB accumulated NLL

  const int tid  = threadIdx.x;
  const int lane = tid & 31;
  const int wv   = tid >> 5;              // wave 0..15, owns columns [32wv, 32wv+32)
  const int wg   = blockIdx.x;            // batch slice
  const int col  = lane & 15;             // N within a 16-wide tile
  const int hi8  = (lane >> 4) << 3;      // D-fragment row offset (0 or 8)

  // ---- init state ----
  for (int i = tid; i < 16 * RNN; i += 512) { h0_bf[i] = (bf16)0.0f; h1_bf[i] = (bf16)0.0f; }
  for (int i = tid; i < 16 * STATE; i += 512) nll_ls[i] = 0.0f;
  for (int i = tid; i < 16 * ZDIM; i += 512) {
    int r = i >> 6, c = i & 63;
    z_bf[i] = (bf16)z[(size_t)(wg * 16 + r) * ZDIM + c];
  }
  __syncthreads();

  for (int t = 0; t < T_STEPS; ++t) {
    // Re-materialized (opaque, global-AS) weight pointers: loads through these
    // cannot be hoisted out of the t-loop, so B-fragments stream from L2.
    gwptr Wfc1  = opq(Wfc1p);
    gwptr Wfc2  = opq(Wfc2p);
    gwptr Wmean = opq(Wmeanp);
    gwptr Wlv   = opq(Wlvp);
    gwptr Wih0  = opq(Wih0p);
    gwptr Whh0  = opq(Whh0p);
    gwptr Wih1  = opq(Wih1p);
    gwptr Whh1  = opq(Whh1p);

    // ---- stage s_t and a_t = states[t+1]-states[t] (512 threads == 512 elems) ----
    {
      int r = tid >> 5, c = tid & 31;
      size_t base = ((size_t)t * BATCH + (wg * 16 + r)) * STATE + c;
      float s0 = states[base];
      float s1 = states[base + (size_t)BATCH * STATE];
      float av = s1 - s0;
      s_bf[tid] = (bf16)s0;
      a_bf[tid] = (bf16)av;
      a_f32[tid] = av;
    }
    __syncthreads();

    // ---- fc1: dh = relu([s|z|h1] @ Wfc1^T + b), K = 608 ----
    #pragma unroll
    for (int tile = 0; tile < 2; ++tile) {
      const int n0 = wv * 32 + tile * 16;
      v8f acc = z8();
      acc = wmma_bf16(load_a_lds(s_bf, STATE, 0, lane),  load_b_w(Wfc1, FCIN, n0, 0, lane),  acc);
      acc = wmma_bf16(load_a_lds(z_bf, ZDIM, 0, lane),   load_b_w(Wfc1, FCIN, n0, 32, lane), acc);
      acc = wmma_bf16(load_a_lds(z_bf, ZDIM, 32, lane),  load_b_w(Wfc1, FCIN, n0, 64, lane), acc);
      for (int kt = 0; kt < 16; ++kt)
        acc = wmma_bf16(load_a_lds(h1_bf, RNN, kt * 32, lane),
                        load_b_w(Wfc1, FCIN, n0, 96 + kt * 32, lane), acc);
      const float bias = bfc1[n0 + col];
      #pragma unroll
      for (int e = 0; e < 8; ++e) {
        float v = acc[e] + bias;
        dh_bf[(e + hi8) * RNN + n0 + col] = (bf16)(v > 0.0f ? v : 0.0f);
      }
    }
    __syncthreads();

    // ---- fc2: dh = relu(dh @ Wfc2^T + b), K = 512 (compute, barrier, overwrite) ----
    {
      v8f accA = z8(), accB = z8();
      const int n0a = wv * 32, n0b = wv * 32 + 16;
      for (int kt = 0; kt < 16; ++kt) {
        v16bf af = load_a_lds(dh_bf, RNN, kt * 32, lane);
        accA = wmma_bf16(af, load_b_w(Wfc2, RNN, n0a, kt * 32, lane), accA);
        accB = wmma_bf16(af, load_b_w(Wfc2, RNN, n0b, kt * 32, lane), accB);
      }
      __syncthreads();            // all reads of dh done before overwrite
      const float biasA = bfc2[n0a + col], biasB = bfc2[n0b + col];
      #pragma unroll
      for (int e = 0; e < 8; ++e) {
        float va = accA[e] + biasA, vb = accB[e] + biasB;
        dh_bf[(e + hi8) * RNN + n0a + col] = (bf16)(va > 0.0f ? va : 0.0f);
        dh_bf[(e + hi8) * RNN + n0b + col] = (bf16)(vb > 0.0f ? vb : 0.0f);
      }
    }
    __syncthreads();

    // ---- heads (waves 0-3) + GRU layer 0 (all waves) ----
    if (wv < 4) {
      gwptr Wm  = (wv < 2) ? Wmean : Wlv;
      const float* bm  = (wv < 2) ? bmean : blv;
      float*       dst = (wv < 2) ? mean_ls : lv_ls;
      const int n0 = (wv & 1) * 16;
      v8f acc = z8();
      for (int kt = 0; kt < 16; ++kt)
        acc = wmma_bf16(load_a_lds(dh_bf, RNN, kt * 32, lane),
                        load_b_w(Wm, RNN, n0, kt * 32, lane), acc);
      const float bias = bm[n0 + col];
      #pragma unroll
      for (int e = 0; e < 8; ++e)
        dst[(e + hi8) * STATE + n0 + col] = acc[e] + bias;
    }

    float h0new[16];
    #pragma unroll
    for (int tile = 0; tile < 2; ++tile) {
      const int j0 = wv * 32 + tile * 16;
      v8f gr = z8(), gz = z8(), gn = z8(), hr = z8(), hz = z8(), hn = z8();
      // gi = [s|a] @ Wih0^T (K = 64)
      {
        v16bf a0 = load_a_lds(s_bf, STATE, 0, lane);
        gr = wmma_bf16(a0, load_b_w(Wih0, 64, j0, 0, lane), gr);
        gz = wmma_bf16(a0, load_b_w(Wih0, 64, RNN + j0, 0, lane), gz);
        gn = wmma_bf16(a0, load_b_w(Wih0, 64, 2 * RNN + j0, 0, lane), gn);
        v16bf a1 = load_a_lds(a_bf, STATE, 0, lane);
        gr = wmma_bf16(a1, load_b_w(Wih0, 64, j0, 32, lane), gr);
        gz = wmma_bf16(a1, load_b_w(Wih0, 64, RNN + j0, 32, lane), gz);
        gn = wmma_bf16(a1, load_b_w(Wih0, 64, 2 * RNN + j0, 32, lane), gn);
      }
      // gh = h0 @ Whh0^T (K = 512)
      for (int kt = 0; kt < 16; ++kt) {
        v16bf ah = load_a_lds(h0_bf, RNN, kt * 32, lane);
        hr = wmma_bf16(ah, load_b_w(Whh0, RNN, j0, kt * 32, lane), hr);
        hz = wmma_bf16(ah, load_b_w(Whh0, RNN, RNN + j0, kt * 32, lane), hz);
        hn = wmma_bf16(ah, load_b_w(Whh0, RNN, 2 * RNN + j0, kt * 32, lane), hn);
      }
      const float bir = bih0[j0 + col], biz = bih0[RNN + j0 + col], bin_ = bih0[2 * RNN + j0 + col];
      const float bhr = bhh0[j0 + col], bhz = bhh0[RNN + j0 + col], bhn = bhh0[2 * RNN + j0 + col];
      #pragma unroll
      for (int e = 0; e < 8; ++e) {
        float r  = sigf(gr[e] + bir + hr[e] + bhr);
        float zg = sigf(gz[e] + biz + hz[e] + bhz);
        float nn = tanhf(gn[e] + bin_ + r * (hn[e] + bhn));
        float ho = (float)h0_bf[(e + hi8) * RNN + j0 + col];
        h0new[tile * 8 + e] = (1.0f - zg) * nn + zg * ho;
      }
    }
    __syncthreads();              // all reads of h0 done; mean/lv visible

    // ---- commit h0; wave 0 accumulates NLL ----
    #pragma unroll
    for (int tile = 0; tile < 2; ++tile) {
      const int j0 = wv * 32 + tile * 16;
      #pragma unroll
      for (int e = 0; e < 8; ++e)
        h0_bf[(e + hi8) * RNN + j0 + col] = (bf16)h0new[tile * 8 + e];
    }
    if (wv == 0) {
      const int c = lane;  // 0..31
      for (int r = 0; r < 16; ++r) {
        float av = a_f32[r * STATE + c];
        float m  = mean_ls[r * STATE + c];
        float lv = lv_ls[r * STATE + c];
        float d  = av - m;
        nll_ls[r * STATE + c] += 0.5f * (d * d * __expf(-lv) + lv + LOG2PI_F);
      }
    }
    __syncthreads();              // h0 committed

    // ---- GRU layer 1: input h0(new), hidden h1 ----
    float h1new[16];
    #pragma unroll
    for (int tile = 0; tile < 2; ++tile) {
      const int j0 = wv * 32 + tile * 16;
      v8f gr = z8(), gz = z8(), gn = z8(), hr = z8(), hz = z8(), hn = z8();
      for (int kt = 0; kt < 16; ++kt) {
        const int k0 = kt * 32;
        v16bf ax = load_a_lds(h0_bf, RNN, k0, lane);
        gr = wmma_bf16(ax, load_b_w(Wih1, RNN, j0, k0, lane), gr);
        gz = wmma_bf16(ax, load_b_w(Wih1, RNN, RNN + j0, k0, lane), gz);
        gn = wmma_bf16(ax, load_b_w(Wih1, RNN, 2 * RNN + j0, k0, lane), gn);
        v16bf ah = load_a_lds(h1_bf, RNN, k0, lane);
        hr = wmma_bf16(ah, load_b_w(Whh1, RNN, j0, k0, lane), hr);
        hz = wmma_bf16(ah, load_b_w(Whh1, RNN, RNN + j0, k0, lane), hz);
        hn = wmma_bf16(ah, load_b_w(Whh1, RNN, 2 * RNN + j0, k0, lane), hn);
      }
      const float bir = bih1[j0 + col], biz = bih1[RNN + j0 + col], bin_ = bih1[2 * RNN + j0 + col];
      const float bhr = bhh1[j0 + col], bhz = bhh1[RNN + j0 + col], bhn = bhh1[2 * RNN + j0 + col];
      #pragma unroll
      for (int e = 0; e < 8; ++e) {
        float r  = sigf(gr[e] + bir + hr[e] + bhr);
        float zg = sigf(gz[e] + biz + hz[e] + bhz);
        float nn = tanhf(gn[e] + bin_ + r * (hn[e] + bhn));
        float ho = (float)h1_bf[(e + hi8) * RNN + j0 + col];
        h1new[tile * 8 + e] = (1.0f - zg) * nn + zg * ho;
      }
    }
    __syncthreads();              // all reads of h1 done
    #pragma unroll
    for (int tile = 0; tile < 2; ++tile) {
      const int j0 = wv * 32 + tile * 16;
      #pragma unroll
      for (int e = 0; e < 8; ++e)
        h1_bf[(e + hi8) * RNN + j0 + col] = (bf16)h1new[tile * 8 + e];
    }
    __syncthreads();              // step complete
  }

  // ---- write NLL out: (B, 32) f32 ----
  {
    int r = tid >> 5, c = tid & 31;
    out[(size_t)(wg * 16 + r) * STATE + c] = nll_ls[r * STATE + c];
  }
}

// ---------------- host launcher ----------------
extern "C" void kernel_launch(void* const* d_in, const int* in_sizes, int n_in,
                              void* d_out, int out_size, void* d_ws, size_t ws_size,
                              hipStream_t stream) {
  (void)in_sizes; (void)n_in; (void)out_size; (void)ws_size;
  const float* states = (const float*)d_in[0];
  const float* z      = (const float*)d_in[1];
  const float* bih0   = (const float*)d_in[4];
  const float* bhh0   = (const float*)d_in[5];
  const float* bih1   = (const float*)d_in[8];
  const float* bhh1   = (const float*)d_in[9];
  const float* bfc1   = (const float*)d_in[11];
  const float* bfc2   = (const float*)d_in[13];
  const float* bmean  = (const float*)d_in[15];
  const float* blv    = (const float*)d_in[17];

  bf16* ws = (bf16*)d_ws;
  auto cvt = [&](int idx, size_t off, int n) {
    cvt_f32_to_bf16<<<dim3((n + 255) / 256), dim3(256), 0, stream>>>(
        (const float*)d_in[idx], ws + off, n);
  };
  cvt(2,  O_WIH0,  1536 * 64);    // Wih0
  cvt(3,  O_WHH0,  1536 * 512);   // Whh0
  cvt(6,  O_WIH1,  1536 * 512);   // Wih1
  cvt(7,  O_WHH1,  1536 * 512);   // Whh1
  cvt(10, O_WFC1,  512 * 608);    // Wfc1
  cvt(12, O_WFC2,  512 * 512);    // Wfc2
  cvt(14, O_WMEAN, 32 * 512);     // Wmean
  cvt(16, O_WLV,   32 * 512);     // Wlv

  tvae_decoder_kernel<<<dim3(32), dim3(512), 0, stream>>>(
      states, z,
      ws + O_WIH0, ws + O_WHH0, bih0, bhh0,
      ws + O_WIH1, ws + O_WHH1, bih1, bhh1,
      ws + O_WFC1, bfc1, ws + O_WFC2, bfc2,
      ws + O_WMEAN, bmean, ws + O_WLV, blv,
      (float*)d_out);
}
